// EQGATEdgeGNN_83700322665128
// MI455X (gfx1250) — compile-verified, matrix operand order checked by reading, not run on hardware
//
#include <hip/hip_runtime.h>

// ---------------- problem constants ----------------
#define NN   8192
#define EE   131072
#define SD   256
#define VD   64
#define ED   128
#define DIN  644          // 2*SD + ED + 4
#define DOUT 513          // SD + 2*VD + 1 + ED
#define HID  256
#define NL   5

// ---------------- WMMA / TDM types ----------------
typedef __attribute__((ext_vector_type(16))) __bf16 v16bf;
typedef __attribute__((ext_vector_type(8)))  float  v8f;
typedef __attribute__((ext_vector_type(4)))  unsigned int v4u;
typedef __attribute__((ext_vector_type(8)))  int          v8i;
typedef __attribute__((ext_vector_type(4)))  int          v4i;

struct U128 { unsigned int a, b, c, d; };

#ifndef __has_builtin
#define __has_builtin(x) 0
#endif
#if __has_builtin(__builtin_amdgcn_tensor_load_to_lds) && __has_builtin(__builtin_amdgcn_s_wait_tensorcnt)
#define HAVE_TDM 1
#else
#define HAVE_TDM 0
#endif

static __device__ __forceinline__ unsigned short f2bf(float f) {
    union { __bf16 b; unsigned short u; } c;
    c.b = (__bf16)f;                 // native cvt if available
    return c.u;
}
static __device__ __forceinline__ float silu_f(float x) {
    return x / (1.0f + __expf(-x));
}

// load a 16-bf16 A/B fragment: two contiguous 16B chunks from LDS
static __device__ __forceinline__ v16bf load_frag(const unsigned short* base, int off) {
    union { v16bf v; U128 q[2]; } f;
    f.q[0] = *reinterpret_cast<const U128*>(base + off);
    f.q[1] = *reinterpret_cast<const U128*>(base + off + 16);
    return f.v;
}

// ---------------- GEMM config ----------------
#define BM 128
#define BN 64
#define BK 32
#define LDA_S 40   // padded LDS stride (ushorts): 64B data + 16B pad per row
#define LDB_S 40

enum { AM_FEAT = 0, AM_BF16 = 2 };
enum { EPI_SILU_BF16 = 0, EPI_SCATTER = 1, EPI_ADD_F32 = 2 };

struct GP {
    int M, N, K;
    // A sources (bf16)
    const unsigned short* Ab; int lda;      // AM_BF16 (row-major, K mult of 32)
    // AM_FEAT gather sources
    const unsigned short* slnb; const unsigned short* eb;
    const float* dd; const float* aa; const float* pn;
    const int* src; const int* tgt;
    // B: bf16 row-major K x N
    const unsigned short* Bb; int ldb;
    const float* bias;
    // outputs
    unsigned short* Cb;  // bf16 out
    float* Cf;           // f32 accumulate out
    int ldc;
    // scatter epilogue
    float* s_acc; float* v_acc; float* p_acc;
    const float* vcur; const float* rn;
    float* e_f32; unsigned short* e_b16;
    int has_v;
};

#if HAVE_TDM
// Issue a TDM 2D bf16 tile load (tile_w x tile_h elems) into LDS with row padding
// matching LDA_S (64B data + 16B pad). Descriptor packing per CDNA5 ISA section 8.
// clang-23 toolchain: 6-arg builtin (g0 v4u, g1 v8i, v4i, v4i, v8i, i32 cpol).
static __device__ __forceinline__ void tdm_load_tile_bf16(
        const unsigned short* gsrc, unsigned lds_byte_off,
        unsigned tensor_w, unsigned tensor_h, unsigned tile_w, unsigned tile_h) {
    unsigned long long ga = (unsigned long long)gsrc;
    v4u g0;
    g0.x = 1u;                                                // count=1, load, user mode
    g0.y = lds_byte_off;                                      // lds_addr
    g0.z = (unsigned)(ga & 0xFFFFFFFFu);                      // global_addr[31:0]
    g0.w = (unsigned)((ga >> 32) & 0x01FFFFFFu) | (2u << 30); // global_addr[56:32] | type=2
    v8i g1;
    // data_size=1 (2B), pad_enable, pad_interval=3 (64B), pad_amount=3 (4 dwords = 16B)
    g1[0] = (1 << 16) | (1 << 20) | (3 << 22) | (3 << 25);
    g1[1] = (int)((tensor_w & 0xFFFFu) << 16);                // tensor_dim0[15:0] -> [63:48]
    g1[2] = (int)(((tensor_w >> 16) & 0xFFFFu) | ((tensor_h & 0xFFFFu) << 16));
    g1[3] = (int)(((tensor_h >> 16) & 0xFFFFu) | ((tile_w & 0xFFFFu) << 16));
    g1[4] = (int)(tile_h & 0xFFFFu);                          // tile_dim1 (tile_dim2=0)
    g1[5] = (int)tensor_w;                                    // tensor_dim0_stride[31:0]
    g1[6] = 0;
    g1[7] = 0;
    v4i z4 = {0, 0, 0, 0};
    v8i z8 = {0, 0, 0, 0, 0, 0, 0, 0};
    __builtin_amdgcn_tensor_load_to_lds(g0, g1, z4, z4, z8, 0);
}
#endif

template <int AMODE, int EPI>
__global__ __launch_bounds__(256) void gemm_wmma(GP g) {
    __shared__ unsigned short At[2][BM * LDA_S];
    __shared__ unsigned short Bt[2][BN * LDB_S];

    const int tid  = threadIdx.x;
    const int lane = tid & 31;
    const int wave = tid >> 5;
    const int wm   = (wave & 3) * 32;     // 4 waves along M
    const int wn   = (wave >> 2) * 32;    // 2 waves along N
    const int l16  = lane & 15;
    const int lh   = (lane >> 4) & 1;
    const int m0   = blockIdx.x * BM;
    const int n0   = blockIdx.y * BN;

#if HAVE_TDM
    constexpr bool TDM_A = (AMODE == AM_BF16);
#else
    constexpr bool TDM_A = false;
#endif

    v8f acc[2][2];
#pragma unroll
    for (int i = 0; i < 2; ++i)
#pragma unroll
        for (int j = 0; j < 2; ++j)
#pragma unroll
            for (int r = 0; r < 8; ++r) acc[i][j][r] = 0.0f;

    // A-fill assignment: 2 threads per row, 16 contiguous ks each (bf16 copies)
    const int ar   = tid >> 1;
    const int ak0  = (tid & 1) * 16;
    const int gm_a = m0 + ar;
    int a_src = 0, a_tgt = 0;
    if (AMODE == AM_FEAT) { a_src = g.src[gm_a]; a_tgt = g.tgt[gm_a]; }

    // B-fill assignment: thread -> column (tid&63), 8 consecutive ks
    const int bn   = tid & 63;
    const int bk0  = (tid >> 6) * 8;
    const int gn_b = n0 + bn;

    const int KT = (g.K + BK - 1) / BK;

    // register-staged tiles
    U128 abf[2];
    unsigned short bus[8];

    auto load_tile = [&](int kt) {
        const int k0 = kt * BK;
        const int kk = k0 + ak0;
        // ---- A (skipped when TDM drives the A panel) ----
        if (!TDM_A) {
            if (AMODE == AM_BF16) {
                const U128* p = reinterpret_cast<const U128*>(g.Ab + (size_t)gm_a * g.lda + kk);
                abf[0] = p[0];
                abf[1] = p[1];
            } else { // AM_FEAT: region uniform per 16-wide chunk (boundaries 16-aligned)
                const unsigned short* p = nullptr;
                if (kk + 16 <= SD)                               p = g.slnb + (size_t)a_src * SD + kk;
                else if (kk >= SD && kk + 16 <= 2 * SD)          p = g.slnb + (size_t)a_tgt * SD + (kk - SD);
                else if (kk >= 2 * SD && kk + 16 <= 2 * SD + ED) p = g.eb   + (size_t)gm_a * ED + (kk - 2 * SD);
                if (p) {
                    const U128* p4 = reinterpret_cast<const U128*>(p);
                    abf[0] = p4[0];
                    abf[1] = p4[1];
                } else {
                    union { U128 q[2]; unsigned short u[16]; } t;
#pragma unroll
                    for (int j = 0; j < 16; ++j) t.u[j] = 0;
                    if (kk == 2 * SD + ED) {   // scalar tail, k = 640..643
                        t.u[0] = f2bf(g.dd[gm_a]);
                        t.u[1] = f2bf(g.aa[gm_a]);
                        t.u[2] = f2bf(g.pn[a_src]);
                        t.u[3] = f2bf(g.pn[a_tgt]);
                    }
                    abf[0] = t.q[0]; abf[1] = t.q[1];
                }
            }
        }
        // ---- B (bf16 strided column reads; coalesced across threads) ----
#pragma unroll
        for (int j = 0; j < 8; ++j) {
            const int gk = k0 + bk0 + j;
            bus[j] = (gk < g.K && gn_b < g.N) ? g.Bb[(size_t)gk * g.ldb + gn_b] : (unsigned short)0;
        }
        const int pk = k0 + 2 * BK + bk0;
        if (pk < g.K && gn_b < g.N)
            __builtin_prefetch(&g.Bb[(size_t)pk * g.ldb + gn_b], 0, 3);
    };

    auto store_tile = [&](int buf) {
        if (!TDM_A) {
            *reinterpret_cast<U128*>(&At[buf][ar * LDA_S + ak0 + 0]) = abf[0];
            *reinterpret_cast<U128*>(&At[buf][ar * LDA_S + ak0 + 8]) = abf[1];
        }
        union { U128 q; unsigned short u[8]; } pb;
#pragma unroll
        for (int j = 0; j < 8; ++j) pb.u[j] = bus[j];
        *reinterpret_cast<U128*>(&Bt[buf][bn * LDB_S + bk0]) = pb.q;
    };

#if HAVE_TDM
    auto tdm_issue = [&](int kt, int buf) {
        if (wave == 0) {
            const unsigned short* src = g.Ab + (size_t)m0 * g.lda + kt * BK;
            unsigned lds_off = (unsigned)(size_t)&At[buf][0];
            tdm_load_tile_bf16(src, lds_off, (unsigned)g.lda, (unsigned)g.M, BK, BM);
        }
    };
#endif

#if HAVE_TDM
    if (TDM_A) tdm_issue(0, 0);
#endif
    load_tile(0);
    store_tile(0);
#if HAVE_TDM
    if (TDM_A && wave == 0) __builtin_amdgcn_s_wait_tensorcnt(0);
#endif
    __syncthreads();

    for (int kt = 0; kt < KT; ++kt) {
        const int cur = kt & 1;
        const int nxt = cur ^ 1;
        if (kt + 1 < KT) {
#if HAVE_TDM
            if (TDM_A) tdm_issue(kt + 1, nxt);   // async A-panel DMA overlaps WMMA
#endif
            load_tile(kt + 1);
        }

        const unsigned short* Ac = At[cur];
        const unsigned short* Bc = Bt[cur];
        v16bf af0 = load_frag(Ac, (wm + 0  + l16) * LDA_S + lh * 8);
        v16bf af1 = load_frag(Ac, (wm + 16 + l16) * LDA_S + lh * 8);
        v16bf bf0 = load_frag(Bc, (wn + 0  + l16) * LDB_S + lh * 8);
        v16bf bf1 = load_frag(Bc, (wn + 16 + l16) * LDB_S + lh * 8);

        acc[0][0] = __builtin_amdgcn_wmma_f32_16x16x32_bf16(false, af0, false, bf0, (short)0, acc[0][0], false, false);
        acc[0][1] = __builtin_amdgcn_wmma_f32_16x16x32_bf16(false, af0, false, bf1, (short)0, acc[0][1], false, false);
        acc[1][0] = __builtin_amdgcn_wmma_f32_16x16x32_bf16(false, af1, false, bf0, (short)0, acc[1][0], false, false);
        acc[1][1] = __builtin_amdgcn_wmma_f32_16x16x32_bf16(false, af1, false, bf1, (short)0, acc[1][1], false, false);

        if (kt + 1 < KT) store_tile(nxt);
#if HAVE_TDM
        if (TDM_A && wave == 0 && kt + 1 < KT) __builtin_amdgcn_s_wait_tensorcnt(0);
#endif
        __syncthreads();
    }

    // ---- epilogue ----
#pragma unroll
    for (int sm = 0; sm < 2; ++sm) {
#pragma unroll
        for (int sn = 0; sn < 2; ++sn) {
#pragma unroll
            for (int r = 0; r < 8; ++r) {
                const int gm = m0 + wm + sm * 16 + lh * 8 + r;
                const int gn = n0 + wn + sn * 16 + l16;
                if (gn >= g.N) continue;
                float val = acc[sm][sn][r];
                if (g.bias) val += g.bias[gn];

                if (EPI == EPI_SILU_BF16) {
                    g.Cb[(size_t)gm * g.ldc + gn] = f2bf(silu_f(val));
                } else if (EPI == EPI_ADD_F32) {
                    g.Cf[(size_t)gm * g.ldc + gn] += val;
                } else { // EPI_SCATTER
                    const int t = g.tgt[gm];
                    if (gn < SD) {
                        atomicAdd(&g.s_acc[(size_t)t * SD + gn], val);
                    } else if (gn < SD + VD) {            // wv1 * v[src]
                        if (g.has_v) {
                            const int s = g.src[gm];
                            const int j = gn - SD;
#pragma unroll
                            for (int c = 0; c < 3; ++c)
                                atomicAdd(&g.v_acc[(size_t)t * (3 * VD) + c * VD + j],
                                          val * g.vcur[(size_t)s * (3 * VD) + c * VD + j]);
                        }
                    } else if (gn < SD + 2 * VD) {        // wv2 * rn
                        const int j = gn - (SD + VD);
#pragma unroll
                        for (int c = 0; c < 3; ++c)
                            atomicAdd(&g.v_acc[(size_t)t * (3 * VD) + c * VD + j],
                                      val * g.rn[(size_t)gm * 3 + c]);
                    } else if (gn == SD + 2 * VD) {       // tanh(wp) * rn
                        const float th = tanhf(val);
#pragma unroll
                        for (int c = 0; c < 3; ++c)
                            atomicAdd(&g.p_acc[(size_t)t * 3 + c], th * g.rn[(size_t)gm * 3 + c]);
                    } else {                              // e_new -> silu (f32 + bf16 mirror)
                        const float r2 = silu_f(val);
                        g.e_f32[(size_t)gm * ED + (gn - (SD + 2 * VD + 1))] = r2;
                        g.e_b16[(size_t)gm * ED + (gn - (SD + 2 * VD + 1))] = f2bf(r2);
                    }
                }
            }
        }
    }
}

// ---------------- elementwise kernels ----------------
__global__ void cvt_bf16_kernel(const float* in, unsigned short* out, long long n) {
    long long i = (long long)blockIdx.x * blockDim.x + threadIdx.x;
    if (i < n) out[i] = f2bf(in[i]);
}

__global__ void count_kernel(const int* tgt, float* cnt, int e) {
    int i = blockIdx.x * blockDim.x + threadIdx.x;
    if (i < e) atomicAdd(&cnt[tgt[i]], 1.0f);
}

__global__ void edge_attr_kernel(const int* src, const int* tgt, const float* p,
                                 float* d, float* a, float* rn, int e) {
    int i = blockIdx.x * blockDim.x + threadIdx.x;
    if (i >= e) return;
    const int s = src[i], t = tgt[i];
    float ps0 = p[s * 3 + 0], ps1 = p[s * 3 + 1], ps2 = p[s * 3 + 2];
    float pt0 = p[t * 3 + 0], pt1 = p[t * 3 + 1], pt2 = p[t * 3 + 2];
    float r0 = pt0 - ps0, r1 = pt1 - ps1, r2 = pt2 - ps2;
    float rr = r0 * r0 + r1 * r1 + r2 * r2;
    float dd = sqrtf(fmaxf(rr, 1e-6f));
    float inv = 1.0f / (1.0f + dd);
    d[i] = dd;
    a[i] = pt0 * ps0 + pt1 * ps1 + pt2 * ps2;
    rn[i * 3 + 0] = r0 * inv;
    rn[i * 3 + 1] = r1 * inv;
    rn[i * 3 + 2] = r2 * inv;
}

__global__ void pn_kernel(const float* p, float* pn, int n) {
    int i = blockIdx.x * blockDim.x + threadIdx.x;
    if (i >= n) return;
    float x = p[i * 3 + 0], y = p[i * 3 + 1], z = p[i * 3 + 2];
    pn[i] = sqrtf(x * x + y * y + z * z);
}

__global__ void ln_kernel(const float* s, const float* g, const float* b, unsigned short* out) {
    __shared__ float red[SD];
    const int row = blockIdx.x, t = threadIdx.x;
    float v = s[(size_t)row * SD + t];
    red[t] = v; __syncthreads();
    for (int o = SD / 2; o > 0; o >>= 1) { if (t < o) red[t] += red[t + o]; __syncthreads(); }
    float mu = red[0] / (float)SD; __syncthreads();
    float dv = v - mu;
    red[t] = dv * dv; __syncthreads();
    for (int o = SD / 2; o > 0; o >>= 1) { if (t < o) red[t] += red[t + o]; __syncthreads(); }
    float var = red[0] / (float)SD;
    out[(size_t)row * SD + t] = f2bf(g[t] * dv * rsqrtf(var + 1e-5f) + b[t]);
}

__global__ void node_update_kernel(float* s, unsigned short* s_b, float* v, float* p,
                                   const float* sa, const float* va, const float* pa,
                                   const float* cnt, int n) {
    int idx = blockIdx.x * blockDim.x + threadIdx.x;
    const int nS = n * SD, nV = n * 3 * VD;
    if (idx < nS) {
        float x = s[idx] + sa[idx];
        s[idx] = x;
        s_b[idx] = f2bf(x);
    } else if (idx < nS + nV) {
        int j = idx - nS;
        float c = fmaxf(cnt[j / (3 * VD)], 1.0f);
        v[j] += va[j] / c;
    } else if (idx < nS + nV + n * 3) {
        int j = idx - nS - nV;
        float c = fmaxf(cnt[j / 3], 1.0f);
        p[j] += pa[j] / c;
    }
}

// ---------------- host driver ----------------
extern "C" void kernel_launch(void* const* d_in, const int* in_sizes, int n_in,
                              void* d_out, int out_size, void* d_ws, size_t ws_size,
                              hipStream_t stream) {
    (void)in_sizes; (void)n_in; (void)out_size; (void)ws_size;

    const float* s_in = (const float*)d_in[0];
    const float* v_in = (const float*)d_in[1];
    const float* p_in = (const float*)d_in[2];
    const float* e_in = (const float*)d_in[3];
    const int*   ei   = (const int*)  d_in[4];
    const float* ln_g = (const float*)d_in[5];
    const float* ln_b = (const float*)d_in[6];
    const float* W1   = (const float*)d_in[7];
    const float* b1   = (const float*)d_in[8];
    const float* W2   = (const float*)d_in[9];
    const float* b2   = (const float*)d_in[10];
    const float* U1   = (const float*)d_in[11];
    const float* bU1  = (const float*)d_in[12];
    const float* U2   = (const float*)d_in[13];
    const float* bU2  = (const float*)d_in[14];
    const int* src = ei;
    const int* tgt = ei + EE;

    char* ws = (char*)d_ws;
    size_t off = 0;
    auto alloc = [&](size_t bytes) -> char* {
        char* q = ws + off;
        off += (bytes + 255) & ~(size_t)255;
        return q;
    };
    float* s_cur = (float*)alloc((size_t)NN * SD * 4);
    float* v_cur = (float*)alloc((size_t)NN * 3 * VD * 4);
    float* p_cur = (float*)alloc((size_t)NN * 3 * 4);
    float* e_cur = (float*)alloc((size_t)EE * ED * 4);
    float* pn    = (float*)alloc((size_t)NN * 4);
    float* dbuf  = (float*)alloc((size_t)EE * 4);
    float* abuf  = (float*)alloc((size_t)EE * 4);
    float* rn    = (float*)alloc((size_t)EE * 3 * 4);
    float* cnt   = (float*)alloc((size_t)NN * 4);
    float* s_acc = (float*)alloc((size_t)NN * SD * 4);
    float* v_acc = (float*)alloc((size_t)NN * 3 * VD * 4);
    float* p_acc = (float*)alloc((size_t)NN * 3 * 4);
    unsigned short* s_lnb = (unsigned short*)alloc((size_t)NN * SD * 2);
    unsigned short* s_bf  = (unsigned short*)alloc((size_t)NN * SD * 2);
    unsigned short* e_b   = (unsigned short*)alloc((size_t)EE * ED * 2);
    unsigned short* hbuf  = (unsigned short*)alloc((size_t)EE * HID * 2);
    unsigned short* hubuf = (unsigned short*)alloc((size_t)NN * SD * 2);
    unsigned short* w1b   = (unsigned short*)alloc((size_t)NL * DIN * HID * 2);
    unsigned short* w2b   = (unsigned short*)alloc((size_t)NL * HID * DOUT * 2);
    unsigned short* u1b   = (unsigned short*)alloc((size_t)NL * SD * SD * 2);
    unsigned short* u2b   = (unsigned short*)alloc((size_t)NL * SD * SD * 2);

    // ---- init state + bf16 weight pre-conversion ----
    hipMemcpyAsync(s_cur, s_in, (size_t)NN * SD * 4,     hipMemcpyDeviceToDevice, stream);
    hipMemcpyAsync(v_cur, v_in, (size_t)NN * 3 * VD * 4, hipMemcpyDeviceToDevice, stream);
    hipMemcpyAsync(p_cur, p_in, (size_t)NN * 3 * 4,      hipMemcpyDeviceToDevice, stream);
    hipMemcpyAsync(e_cur, e_in, (size_t)EE * ED * 4,     hipMemcpyDeviceToDevice, stream);
    hipMemsetAsync(cnt, 0, (size_t)NN * 4, stream);
    count_kernel<<<EE / 256, 256, 0, stream>>>(tgt, cnt, EE);

    auto cvt = [&](const float* in, unsigned short* out, long long n) {
        cvt_bf16_kernel<<<(unsigned)((n + 255) / 256), 256, 0, stream>>>(in, out, n);
    };
    cvt(e_in, e_b, (long long)EE * ED);
    cvt(W1, w1b, (long long)NL * DIN * HID);
    cvt(W2, w2b, (long long)NL * HID * DOUT);
    cvt(U1, u1b, (long long)NL * SD * SD);
    cvt(U2, u2b, (long long)NL * SD * SD);

    for (int i = 0; i < NL; ++i) {
        pn_kernel<<<NN / 256, 256, 0, stream>>>(p_cur, pn, NN);
        edge_attr_kernel<<<EE / 256, 256, 0, stream>>>(src, tgt, p_cur, dbuf, abuf, rn, EE);
        ln_kernel<<<NN, SD, 0, stream>>>(s_cur, ln_g + i * SD, ln_b + i * SD, s_lnb);

        // GEMM1: feat(E x 644) @ W1 -> h = silu(.) (bf16)
        {
            GP g = {};
            g.M = EE; g.N = HID; g.K = DIN;
            g.slnb = s_lnb; g.eb = e_b; g.dd = dbuf; g.aa = abuf; g.pn = pn;
            g.src = src; g.tgt = tgt;
            g.Bb = w1b + (size_t)i * DIN * HID; g.ldb = HID;
            g.bias = b1 + (size_t)i * HID;
            g.Cb = hbuf; g.ldc = HID;
            dim3 grid(EE / BM, (HID + BN - 1) / BN);
            gemm_wmma<AM_FEAT, EPI_SILU_BF16><<<grid, 256, 0, stream>>>(g);
        }

        hipMemsetAsync(s_acc, 0, (size_t)NN * SD * 4,     stream);
        hipMemsetAsync(v_acc, 0, (size_t)NN * 3 * VD * 4, stream);
        hipMemsetAsync(p_acc, 0, (size_t)NN * 3 * 4,      stream);

        // GEMM2: h(E x 256) @ W2 -> fused scatter + e update (A-panel via TDM)
        {
            GP g = {};
            g.M = EE; g.N = DOUT; g.K = HID;
            g.Ab = hbuf; g.lda = HID;
            g.Bb = w2b + (size_t)i * HID * DOUT; g.ldb = DOUT;
            g.bias = b2 + (size_t)i * DOUT;
            g.src = src; g.tgt = tgt;
            g.s_acc = s_acc; g.v_acc = v_acc; g.p_acc = p_acc;
            g.vcur = v_cur; g.rn = rn;
            g.e_f32 = e_cur; g.e_b16 = e_b;
            g.has_v = (i == 0) ? 0 : 1;
            dim3 grid(EE / BM, (DOUT + BN - 1) / BN);
            gemm_wmma<AM_BF16, EPI_SCATTER><<<grid, 256, 0, stream>>>(g);
        }

        // apply accumulated messages (also refresh bf16 mirror of s)
        {
            int total = NN * (SD + 3 * VD + 3);
            node_update_kernel<<<(total + 255) / 256, 256, 0, stream>>>(
                s_cur, s_bf, v_cur, p_cur, s_acc, v_acc, p_acc, cnt, NN);
        }

        // node MLP: s += silu(s@U1 + bU1) @ U2 + bU2   (layers 0..3)
        if (i < NL - 1) {
            {
                GP g = {};
                g.M = NN; g.N = SD; g.K = SD;
                g.Ab = s_bf; g.lda = SD;
                g.Bb = u1b + (size_t)i * SD * SD; g.ldb = SD;
                g.bias = bU1 + (size_t)i * SD;
                g.Cb = hubuf; g.ldc = SD;
                dim3 grid(NN / BM, SD / BN);
                gemm_wmma<AM_BF16, EPI_SILU_BF16><<<grid, 256, 0, stream>>>(g);
            }
            {
                GP g = {};
                g.M = NN; g.N = SD; g.K = SD;
                g.Ab = hubuf; g.lda = SD;
                g.Bb = u2b + (size_t)i * SD * SD; g.ldb = SD;
                g.bias = bU2 + (size_t)i * SD;
                g.Cf = s_cur; g.ldc = SD;
                dim3 grid(NN / BM, SD / BN);
                gemm_wmma<AM_BF16, EPI_ADD_F32><<<grid, 256, 0, stream>>>(g);
            }
        }
    }

    // ---- pack outputs: (s, v, e, p) ----
    float* out = (float*)d_out;
    size_t o = 0;
    hipMemcpyAsync(out + o, s_cur, (size_t)NN * SD * 4,     hipMemcpyDeviceToDevice, stream); o += (size_t)NN * SD;
    hipMemcpyAsync(out + o, v_cur, (size_t)NN * 3 * VD * 4, hipMemcpyDeviceToDevice, stream); o += (size_t)NN * 3 * VD;
    hipMemcpyAsync(out + o, e_cur, (size_t)EE * ED * 4,     hipMemcpyDeviceToDevice, stream); o += (size_t)EE * ED;
    hipMemcpyAsync(out + o, p_cur, (size_t)NN * 3 * 4,      hipMemcpyDeviceToDevice, stream);
}
